// SparseConv2d_33363305955462
// MI455X (gfx1250) — matrix operand order repaired and checked by reading, not compiled, not run on hardware
//
#include <hip/hip_runtime.h>
#include <hip/hip_bf16.h>
#include <stdint.h>

typedef __attribute__((ext_vector_type(16))) __bf16 v16bf;
typedef __attribute__((ext_vector_type(8)))  float  v8f;

#define C_IN   256
#define C_OUT  256
#define HH     64
#define WW     64
#define NB     16
#define KTAPS  9
#define KC     32                 // K-chunk (channels)
#define MTILE  128                // output channels per workgroup
#define WBLK   12288              // halves per (coblk,cc,kh) swizzled weight block
#define XBLK   2048               // halves per (img,h,cc) activation block (64 w * 32 k)

// fp32 -> bf16 round-to-nearest-even (bit trick)
__device__ __forceinline__ unsigned short f2bf(float f) {
    union { float f; uint32_t u; } x; x.f = f;
    uint32_t u = x.u;
    u += 0x7FFFu + ((u >> 16) & 1u);
    return (unsigned short)(u >> 16);
}

// ---------------------------------------------------------------------------
// Prep 1: swizzle weights into WMMA A-fragment order (bf16).
// Wswz[((coblk*8 + cc)*3 + kh)*12288 + f], f = kw*4096 + cb*512 + lane*16 + e
// A-fragment order used here (16x32, contiguous k, same as B): m = lane&15, K = (lane/16)*16 + e
// ---------------------------------------------------------------------------
__global__ void prep_weights(const float* __restrict__ weight,
                             unsigned short* __restrict__ Wswz) {
    int g = blockIdx.x * 256 + threadIdx.x;        // < 589824
    int f    = g % WBLK;
    int blk  = g / WBLK;                           // 0..47
    int kh   = blk % 3;
    int cc   = (blk / 3) & 7;
    int cob  = blk / 24;
    int e    = f & 15;
    int ln   = (f >> 4) & 31;
    int cb   = (f >> 9) & 7;
    int kw   = f >> 12;
    int m    = ln & 15;
    int K    = ((ln >> 4) << 4) + e;
    int co   = cob * MTILE + (cb << 4) + m;
    int ci   = (cc << 5) + K;
    Wswz[g] = f2bf(weight[co * (C_IN * KTAPS) + ci * KTAPS + kh * 3 + kw]);
}

// ---------------------------------------------------------------------------
// Prep 2: convert x to bf16, transposed per row-chunk:
// Xt[(((img*64 + h)*8 + cc)*64 + w)*32 + k] = bf16(x[img][cc*32+k][h][w])
// ---------------------------------------------------------------------------
__global__ void prep_x(const float* __restrict__ x,
                       unsigned short* __restrict__ Xt) {
    int g   = blockIdx.x * 256 + threadIdx.x;      // < 4,194,304
    int k   = g & 31;
    int w4  = (g >> 5) & 15;
    int cc  = (g >> 9) & 7;
    int h   = (g >> 12) & 63;
    int img = g >> 18;
    int ci  = (cc << 5) + k;
    const float4 v = *reinterpret_cast<const float4*>(
        &x[(((size_t)(img * C_IN + ci) * HH + h) * WW) + (w4 << 2)]);
    size_t ob = ((size_t)((img * HH + h) * 8 + cc) * WW + (w4 << 2)) * KC + k;
    Xt[ob         ] = f2bf(v.x);
    Xt[ob + KC    ] = f2bf(v.y);
    Xt[ob + 2 * KC] = f2bf(v.z);
    Xt[ob + 3 * KC] = f2bf(v.w);
}

// ---------------------------------------------------------------------------
// Main kernel: double-buffered async-LDS staging + bf16 WMMA implicit GEMM.
// Wave tiling: 4 (M) x 2 (N) waves; each wave computes M=32 x N=32 (acc[2][2]).
// ---------------------------------------------------------------------------
__global__ __launch_bounds__(256, 2)
void conv_main(const unsigned short* __restrict__ Wswz,
               const unsigned short* __restrict__ Xt,
               const float* __restrict__ bias,
               float* __restrict__ out) {
    __shared__ unsigned short Wlds[2][WBLK];       // 2 x 24576 B
    __shared__ unsigned short Xlds[2][66 * KC];    // 2 x  4224 B

    const int tid  = threadIdx.x;
    const int wave = tid >> 5;
    const int lane = tid & 31;
    const int mrow = wave >> 1;                    // 0..3  (M group)
    const int ncol = wave & 1;                     // 0..1  (N group)
    const int row  = blockIdx.x;
    const int img  = row >> 6;
    const int h    = row & 63;
    const int cob  = blockIdx.y;
    const int co0  = cob * MTILE;
    const int halfsel = lane >> 4;
    const int nlane   = lane & 15;
    const unsigned int vbyte = tid << 4;           // per-thread 16B slot

    // Halo columns (wi=0, wi=65) are permanent zeros in both buffers.
    if (tid < 32) {
        Xlds[0][tid] = 0; Xlds[0][65 * KC + tid] = 0;
        Xlds[1][tid] = 0; Xlds[1][65 * KC + tid] = 0;
    }
    __syncthreads();

    v8f acc[2][2] = {};

    auto stage = [&](int cc, int kh, int buf) {
        // ---- weights: 24576 B contiguous -> 6 async b128 per thread ----
        {
            unsigned long long wsrc = (unsigned long long)(uintptr_t)
                (Wswz + (size_t)(((cob << 3) + cc) * 3 + kh) * WBLK);
            unsigned int ldsW = (unsigned int)(uintptr_t)(&Wlds[buf][0]) + vbyte;
            asm volatile(
                "global_load_async_to_lds_b128 %0, %1, %2 offset:0\n\t"
                "global_load_async_to_lds_b128 %0, %1, %2 offset:4096\n\t"
                "global_load_async_to_lds_b128 %0, %1, %2 offset:8192\n\t"
                "global_load_async_to_lds_b128 %0, %1, %2 offset:12288\n\t"
                "global_load_async_to_lds_b128 %0, %1, %2 offset:16384\n\t"
                "global_load_async_to_lds_b128 %0, %1, %2 offset:20480"
                :: "v"(ldsW), "v"(vbyte), "s"(wsrc) : "memory");
        }
        // ---- activations: 4096 B contiguous -> 1 async b128 per thread ----
        const int h_in = h + kh - 1;
        if ((unsigned)h_in < (unsigned)HH) {
            unsigned long long xsrc = (unsigned long long)(uintptr_t)
                (Xt + (size_t)((img * HH + h_in) * 8 + cc) * XBLK);
            unsigned int ldsX = (unsigned int)(uintptr_t)(&Xlds[buf][KC]) + vbyte;
            asm volatile(
                "global_load_async_to_lds_b128 %0, %1, %2 offset:0"
                :: "v"(ldsX), "v"(vbyte), "s"(xsrc) : "memory");
        } else {
            uint4 z = {0u, 0u, 0u, 0u};
            *reinterpret_cast<uint4*>(&Xlds[buf][KC + tid * 8]) = z;
        }
    };

    int cc = 0, kh = 0;
    stage(0, 0, 0);

    for (int it = 0; it < 24; ++it) {
        const int buf = it & 1;
        asm volatile("s_wait_asynccnt 0x0" ::: "memory");  // our tile landed
        __syncthreads();                                    // everyone's tile landed

        // advance (cc, kh) incrementally; stage next tile to overlap with WMMA
        int kh2 = kh + 1, cc2 = cc;
        if (kh2 == 3) { kh2 = 0; cc2 = cc + 1; }
        if (it + 1 < 24) stage(cc2, kh2, buf ^ 1);
        kh = kh2; cc = cc2;

        #pragma unroll
        for (int kw = 0; kw < 3; ++kw) {
            const unsigned short* Wb = &Wlds[buf][kw << 12];   // kw*4096
            // grouped fragment loads (8 x ds_load_b128), then 4 WMMAs
            const v16bf a0 = *reinterpret_cast<const v16bf*>(
                &Wb[(((mrow << 1) | 0) << 9) + (lane << 4)]);
            const v16bf a1 = *reinterpret_cast<const v16bf*>(
                &Wb[(((mrow << 1) | 1) << 9) + (lane << 4)]);
            const int w0 = ((ncol << 1) << 4) + kw + nlane;    // 0..65
            const v16bf b0 = *reinterpret_cast<const v16bf*>(
                &Xlds[buf][w0 * KC + (halfsel << 4)]);
            const v16bf b1 = *reinterpret_cast<const v16bf*>(
                &Xlds[buf][(w0 + 16) * KC + (halfsel << 4)]);
            acc[0][0] = __builtin_amdgcn_wmma_f32_16x16x32_bf16(
                false, a0, false, b0, (short)0, acc[0][0], false, false);
            acc[0][1] = __builtin_amdgcn_wmma_f32_16x16x32_bf16(
                false, a0, false, b1, (short)0, acc[0][1], false, false);
            acc[1][0] = __builtin_amdgcn_wmma_f32_16x16x32_bf16(
                false, a1, false, b0, (short)0, acc[1][0], false, false);
            acc[1][1] = __builtin_amdgcn_wmma_f32_16x16x32_bf16(
                false, a1, false, b1, (short)0, acc[1][1], false, false);
        }
    }

    // ---- epilogue: bias + store (C/D layout: VGPR r -> M = r + halfsel*8) ----
    #pragma unroll
    for (int i = 0; i < 2; ++i) {
        const int co_b = co0 + (((mrow << 1) | i) << 4);
        #pragma unroll
        for (int r = 0; r < 8; ++r) {
            const int co = co_b + r + (halfsel << 3);
            const float b = bias[co];
            #pragma unroll
            for (int j = 0; j < 2; ++j) {
                const int w_out = (((ncol << 1) | j) << 4) + nlane;
                out[(((size_t)(img * C_OUT + co)) * HH + h) * WW + w_out] =
                    acc[i][j][r] + b;
            }
        }
    }
}

// ---------------------------------------------------------------------------
// Fallback (self-contained, no workspace).
// ---------------------------------------------------------------------------
__global__ __launch_bounds__(256, 2)
void conv_fallback(const float* __restrict__ x,
                   const float* __restrict__ weight,
                   const float* __restrict__ bias,
                   float* __restrict__ out) {
    __shared__ unsigned short Wl[3 * 8 * 32 * 16];
    __shared__ unsigned short Xl[66 * KC];
    const int tid  = threadIdx.x;
    const int wave = tid >> 5;
    const int lane = tid & 31;
    const int row  = blockIdx.x;
    const int img  = row >> 6;
    const int h    = row & 63;
    const int co0  = blockIdx.y * MTILE;
    const int halfsel = lane >> 4;
    const int nlane   = lane & 15;
    v8f acc[4] = {};
    for (int ci0 = 0; ci0 < C_IN; ci0 += KC) {
        for (int kh = 0; kh < 3; ++kh) {
            __syncthreads();
            #pragma unroll
            for (int itr = 0; itr < 48; ++itr) {
                int f  = tid + itr * 256;
                int e  = f & 15, ln = (f >> 4) & 31, cb = (f >> 9) & 7, kw = f >> 12;
                int m  = ln & 15;
                int K  = ((ln >> 4) << 4) + e;
                Wl[f] = f2bf(weight[(co0 + (cb << 4) + m) * (C_IN * KTAPS) +
                                    (ci0 + K) * KTAPS + kh * 3 + kw]);
            }
            const int h_in = h + kh - 1;
            #pragma unroll
            for (int itr = 0; itr < 9; ++itr) {
                int f = tid + itr * 256;
                if (f < 66 * KC) {
                    int wi = f / KC, k = f - wi * KC, w_in = wi - 1;
                    float v = 0.0f;
                    if ((unsigned)h_in < (unsigned)HH && (unsigned)w_in < (unsigned)WW)
                        v = x[((img * C_IN + (ci0 + k)) * HH + h_in) * WW + w_in];
                    Xl[wi * KC + k] = f2bf(v);
                }
            }
            __syncthreads();
            #pragma unroll
            for (int kw = 0; kw < 3; ++kw) {
                const v16bf a = *reinterpret_cast<const v16bf*>(
                    &Wl[(((kw << 3) + wave) << 9) + (lane << 4)]);
                #pragma unroll
                for (int t = 0; t < 4; ++t) {
                    const v16bf b = *reinterpret_cast<const v16bf*>(
                        &Xl[((t << 4) + kw + nlane) * KC + (halfsel << 4)]);
                    acc[t] = __builtin_amdgcn_wmma_f32_16x16x32_bf16(
                        false, a, false, b, (short)0, acc[t], false, false);
                }
            }
        }
    }
    const int co_w = co0 + (wave << 4);
    #pragma unroll
    for (int r = 0; r < 8; ++r) {
        const int co = co_w + r + (halfsel << 3);
        const float b = bias[co];
        #pragma unroll
        for (int t = 0; t < 4; ++t)
            out[(((size_t)(img * C_OUT + co)) * HH + h) * WW + (t << 4) + nlane] =
                acc[t][r] + b;
    }
}

extern "C" void kernel_launch(void* const* d_in, const int* in_sizes, int n_in,
                              void* d_out, int out_size, void* d_ws, size_t ws_size,
                              hipStream_t stream) {
    const float* x      = (const float*)d_in[0];
    const float* weight = (const float*)d_in[1];
    const float* bias   = (const float*)d_in[2];
    float* out          = (float*)d_out;

    const size_t WSWZ_BYTES = 589824ull * 2ull;        // 1,179,648
    const size_t XT_BYTES   = 16777216ull * 2ull;      // 33,554,432
    dim3 grid(NB * HH, C_OUT / MTILE);                 // (1024, 2)

    if (ws_size >= WSWZ_BYTES + XT_BYTES) {
        unsigned short* Wswz = (unsigned short*)d_ws;
        unsigned short* Xt   = (unsigned short*)((char*)d_ws + WSWZ_BYTES);
        prep_weights<<<2304, 256, 0, stream>>>(weight, Wswz);
        prep_x<<<16384, 256, 0, stream>>>(x, Xt);
        conv_main<<<grid, 256, 0, stream>>>(Wswz, Xt, bias, out);
    } else {
        conv_fallback<<<grid, 256, 0, stream>>>(x, weight, bias, out);
    }
}
